// RAFT_4148938408126
// MI455X (gfx1250) — matrix operand order repaired and checked by reading, not compiled, not run on hardware
//
#include <hip/hip_runtime.h>
#include <math.h>

// ---------------------------------------------------------------------------
// RAFT forward for MI455X (gfx1250, wave32, WMMA).
// Channel-last f16 activations + pre-reordered [O][KH][KW][Cpad] f16 weights
// so every conv is an implicit GEMM whose A/B fragments are contiguous
// global_load_b128 feeding v_wmma_f32_16x16x32_f16 (same codegen shape as the
// correlation-volume GEMM). Stream-only launches, workspace-carved scratch,
// deterministic -> graph-capture safe.
// ---------------------------------------------------------------------------

typedef _Float16 half_t;
typedef __attribute__((ext_vector_type(16))) _Float16 v16h;
typedef __attribute__((ext_vector_type(8)))  float    v8f;

// Problem constants (fixed by the reference).
constexpr int NB   = 2;            // batch
constexpr int CC   = 256;          // feature channels
constexpr int HH   = 48;
constexpr int WW   = 64;
constexpr int PP   = HH * WW;      // 3072 pixels per image
constexpr int NP   = NB * PP;      // 6144 pixels total
constexpr int HD   = 128;          // hidden dim
constexpr int RADI = 4;
constexpr int NTAP = (2 * RADI + 1) * (2 * RADI + 1);  // 81
constexpr int FEAT = 4 * NTAP;                         // 324
constexpr int FEATP = 352;                             // 324 padded to 32
constexpr int NITER = 12;

// ---------------- WMMA fragment index helpers (wave32, 16-bit) -------------
// A (16x32, MxK): lanes 0-15 row M; K interleaved in 8-blocks across halves.
__device__ __forceinline__ int kfragA(int i, int s, int h) {
  return ((i >> 2) << 4) + (h << 3) + ((i & 3) << 1) + s;
}
// Contiguous K order (used by conv for A and B): lanes 0-15 hold K=0..15 (2/VGPR), lanes 16-31 K=16..31.
__device__ __forceinline__ int kfragB(int i, int s, int h) {
  return (h << 4) + (i << 1) + s;
}

// ------------------------------- kernels -----------------------------------

// fmap [NB, CC, PP] f32  ->  [NB, PP, CC] f16 (K-pairs contiguous for GEMM)
__global__ void k_transpose_pc(const float* __restrict__ s, half_t* __restrict__ d) {
  long i = (long)blockIdx.x * blockDim.x + threadIdx.x;
  if (i >= (long)NB * CC * PP) return;
  int p = (int)(i % PP);
  long t = i / PP;
  int c = (int)(t % CC);
  int n = (int)(t / CC);
  d[((long)n * PP + p) * CC + c] = (half_t)s[i];
}

// OIHW f32 -> [Cpout][KH][KW][Cpin] f16, zero padded in o and c.
__global__ void k_wreorder(const float* __restrict__ src, half_t* __restrict__ dst,
                           int Cout, int Cin, int KH, int KW, int Cpin, int Cpout) {
  long i = (long)blockIdx.x * blockDim.x + threadIdx.x;
  long tot = (long)Cpout * KH * KW * Cpin;
  if (i >= tot) return;
  int c = (int)(i % Cpin);
  long t = i / Cpin;
  int kw = (int)(t % KW);
  t /= KW;
  int kh = (int)(t % KH);
  int o = (int)(t / KH);
  float v = 0.f;
  if (o < Cout && c < Cin) v = src[(((long)o * Cin + c) * KH + kh) * KW + kw];
  dst[i] = (half_t)v;
}

// Correlation volume: corr[n,p,q] = (1/16) * sum_c f1[n,c,p] * f2[n,c,q]
__global__ void k_corr_wmma(const half_t* __restrict__ f1t,
                            const half_t* __restrict__ f2t,
                            float* __restrict__ corr, float scale) {
  const int ptiles = PP / 16;  // 192
  int tile = blockIdx.x * 4 + (threadIdx.x >> 5);
  if (tile >= ptiles * ptiles) return;
  int n = blockIdx.y;
  int mt = tile % ptiles, qt = tile / ptiles;
  int lane = threadIdx.x & 31;
  int hg = lane >> 4, lm = lane & 15;
  const half_t* ar = f1t + ((long)n * PP + mt * 16 + lm) * CC;
  const half_t* br = f2t + ((long)n * PP + qt * 16 + lm) * CC;
  v8f acc = {};
  for (int k0 = 0; k0 < CC; k0 += 32) {
    __builtin_prefetch(ar + k0 + 64, 0, 1);
    __builtin_prefetch(br + k0 + 64, 0, 1);
    v16h a, b;
#pragma unroll
    for (int i = 0; i < 8; ++i) {
      int ka = k0 + kfragA(i, 0, hg);
      int kb = k0 + kfragA(i, 0, hg);
      a[2 * i]     = ar[ka];
      a[2 * i + 1] = ar[ka + 1];
      b[2 * i]     = br[kb];
      b[2 * i + 1] = br[kb + 1];
    }
    acc = __builtin_amdgcn_wmma_f32_16x16x32_f16(false, a, false, b, (short)0,
                                                 acc, false, false);
  }
#pragma unroll
  for (int r = 0; r < 8; ++r) {
    int p = mt * 16 + r + 8 * hg;
    int q = qt * 16 + lm;
    corr[((long)n * PP + p) * PP + q] = acc[r] * scale;
  }
}

// 2x2 average pooling for the pyramid:  s [B,h,w] -> d [B,h/2,w/2]
__global__ void k_avgpool2(const float* __restrict__ s, float* __restrict__ d,
                           int B, int h, int w) {
  int h2 = h >> 1, w2 = w >> 1;
  long i = (long)blockIdx.x * blockDim.x + threadIdx.x;
  if (i >= (long)B * h2 * w2) return;
  int x = (int)(i % w2);
  long t = i / w2;
  int y = (int)(t % h2);
  int b = (int)(t / h2);
  const float* sb = s + (long)b * h * w;
  d[i] = 0.25f * (sb[(2 * y) * w + 2 * x]     + sb[(2 * y) * w + 2 * x + 1] +
                  sb[(2 * y + 1) * w + 2 * x] + sb[(2 * y + 1) * w + 2 * x + 1]);
}

// net = tanh(cnet[:, :128]) -> net_f (f32, [NP,128]) and net128 (f16, [NP,128])
// inp = relu(cnet[:, 128:]) -> written straight into xcat channels [0,128)
__global__ void k_cnet_split(const float* __restrict__ cnet,
                             float* __restrict__ net_f,
                             half_t* __restrict__ net128,
                             half_t* __restrict__ xcat) {
  long i = (long)blockIdx.x * blockDim.x + threadIdx.x;
  if (i >= (long)NP * HD) return;
  int c = (int)(i % HD);
  long np = i / HD;
  int n = (int)(np / PP), p = (int)(np % PP);
  float nv = tanhf(cnet[((long)n * 256 + c) * PP + p]);
  float iv = cnet[((long)n * 256 + HD + c) * PP + p];
  iv = iv > 0.f ? iv : 0.f;
  net_f[i] = nv;
  net128[i] = (half_t)nv;
  xcat[np * 256 + c] = (half_t)iv;
}

// coords1 = coords0 + flow_init   (c1 kept channel-first [NB,2,PP] f32)
__global__ void k_init_c1(const float* __restrict__ flow_init, float* __restrict__ c1) {
  long i = (long)blockIdx.x * blockDim.x + threadIdx.x;
  if (i >= (long)NB * 2 * PP) return;
  int p = (int)(i % PP);
  long t = i / PP;
  int ch = (int)(t % 2);
  float base = (ch == 0) ? (float)(p % WW) : (float)(p / WW);
  c1[i] = base + flow_init[i];
}

// Bilinear pyramid lookup -> feat channel-last [NP, 352] f16 (pad zeroed)
__global__ void k_corr_lookup(const float* __restrict__ pyr0, const float* __restrict__ pyr1,
                              const float* __restrict__ pyr2, const float* __restrict__ pyr3,
                              const float* __restrict__ c1, half_t* __restrict__ feat) {
  long i = (long)blockIdx.x * blockDim.x + threadIdx.x;
  if (i >= (long)NP * FEATP) return;
  int f = (int)(i % FEATP);
  long b = i / FEATP;
  if (f >= FEAT) { feat[i] = (half_t)0.f; return; }
  int n = (int)(b / PP), pq = (int)(b % PP);
  float cx = c1[((long)n * 2) * PP + pq];
  float cy = c1[((long)n * 2 + 1) * PP + pq];
  int lvl = f / NTAP, tap = f % NTAP;
  float inv = 1.0f / (float)(1 << lvl);
  float fx = cx * inv + (float)(tap / 9 - RADI);   // dg[:,0] -> x (reference order)
  float fy = cy * inv + (float)(tap % 9 - RADI);   // dg[:,1] -> y
  int h = HH >> lvl, w = WW >> lvl;
  const float* img =
      (lvl == 0 ? pyr0 : lvl == 1 ? pyr1 : lvl == 2 ? pyr2 : pyr3) + b * (long)h * w;
  float x0 = floorf(fx), y0 = floorf(fy);
  float wx = fx - x0, wy = fy - y0;
  float acc = 0.f;
#pragma unroll
  for (int t2 = 0; t2 < 4; ++t2) {
    float xc = x0 + (float)(t2 & 1), yc = y0 + (float)(t2 >> 1);
    float wgt = ((t2 & 1) ? wx : 1.f - wx) * ((t2 >> 1) ? wy : 1.f - wy);
    if (xc >= 0.f && xc <= (float)(w - 1) && yc >= 0.f && yc <= (float)(h - 1))
      acc += img[(int)yc * w + (int)xc] * wgt;
  }
  feat[i] = (half_t)acc;
}

// flow (f16, channel-last [NP,32], pad zeroed) = coords1 - coords0
__global__ void k_flow_h(const float* __restrict__ c1, half_t* __restrict__ fh) {
  long i = (long)blockIdx.x * blockDim.x + threadIdx.x;
  if (i >= (long)NP * 32) return;
  int c = (int)(i & 31);
  long np = i >> 5;
  float v = 0.f;
  if (c < 2) {
    int n = (int)(np / PP), p = (int)(np % PP);
    float base = (c == 0) ? (float)(p % WW) : (float)(p / WW);
    v = c1[((long)n * 2 + c) * PP + p] - base;
  }
  fh[i] = (half_t)v;
}

// write flow into xcat channels 254/255 (after the mot conv zeroed them)
__global__ void k_flowpair(const float* __restrict__ c1, half_t* __restrict__ xcat) {
  long i = (long)blockIdx.x * blockDim.x + threadIdx.x;
  if (i >= (long)NP * 2) return;
  int ch = (int)(i & 1);
  long np = i >> 1;
  int n = (int)(np / PP), p = (int)(np % PP);
  float base = (ch == 0) ? (float)(p % WW) : (float)(p / WW);
  xcat[np * 256 + 254 + ch] = (half_t)(c1[((long)n * 2 + ch) * PP + p] - base);
}

// Implicit-GEMM conv on WMMA, channel-last activations.
// in : [NP, Cpin] f16        wt : [Cpout][KH][KW][Cpin] f16 (zero padded)
// out: [NP, Cos] (f32 and/or f16) written at channel offset coff; padded
//      output channels (m >= Cout) are written as zero.
__global__ void k_conv_wmma(const half_t* __restrict__ in, const half_t* __restrict__ wt,
                            const float* __restrict__ bias,
                            float* __restrict__ out_f, half_t* __restrict__ out_h,
                            int Cpin, int Cout, int Cpout,
                            int KH, int KW, int PH, int PW,
                            int Cos, int coff, int act, float scale) {
  const int Kp = KH * KW * Cpin;
  const int mtiles = Cpout >> 4;
  const int ptiles = NP >> 4;
  int tile = blockIdx.x * 4 + (threadIdx.x >> 5);
  if (tile >= mtiles * ptiles) return;               // wave-uniform guard
  int mt = tile % mtiles, pt = tile / mtiles;
  int lane = threadIdx.x & 31;
  int hg = lane >> 4, lm = lane & 15;
  int arow = mt * 16 + lm;                            // padded output channel
  int bcol = pt * 16 + lm;                            // global pixel
  int n = bcol / PP, pq = bcol % PP;
  int py = pq / WW, px = pq % WW;
  const half_t* wrow0 = wt + (long)arow * Kp;

  v8f acc = {};
  for (int kh = 0; kh < KH; ++kh) {
    int iy = py + kh - PH;
    for (int kw = 0; kw < KW; ++kw) {
      int ix = px + kw - PW;
      bool valid = (iy >= 0) & (iy < HH) & (ix >= 0) & (ix < WW);
      const half_t* bp = in + ((long)n * PP + iy * WW + ix) * Cpin;
      const half_t* wr = wrow0 + (kh * KW + kw) * Cpin;
      for (int c0 = 0; c0 < Cpin; c0 += 32) {
        __builtin_prefetch(wr + c0 + 64, 0, 1);
        v16h a, b = {};
#pragma unroll
        for (int i = 0; i < 8; ++i) {
          int ka = c0 + kfragB(i, 0, hg);
          a[2 * i]     = wr[ka];
          a[2 * i + 1] = wr[ka + 1];
        }
        if (valid) {
          int kb0 = c0 + (hg << 4);
#pragma unroll
          for (int i = 0; i < 8; ++i) {
            b[2 * i]     = bp[kb0 + 2 * i];
            b[2 * i + 1] = bp[kb0 + 2 * i + 1];
          }
        }
        acc = __builtin_amdgcn_wmma_f32_16x16x32_f16(false, a, false, b, (short)0,
                                                     acc, false, false);
      }
    }
  }
#pragma unroll
  for (int r = 0; r < 8; ++r) {
    int m = mt * 16 + r + 8 * hg;
    float v = 0.f;
    if (m < Cout) {
      v = (acc[r] + bias[m]) * scale;
      if (act == 1)      v = v > 0.f ? v : 0.f;          // relu
      else if (act == 2) v = 1.f / (1.f + expf(-v));     // sigmoid
      else if (act == 3) v = tanhf(v);                   // tanh
    }
    long idx = (long)bcol * Cos + coff + m;
    if (out_f) out_f[idx] = v;
    if (out_h) out_h[idx] = (half_t)v;
  }
}

// channel-last 2-way concat: d[np][ct] = [s0(:c0, stride p0), s1(:c1c, stride p1)]
__global__ void k_concat2(const half_t* __restrict__ s0, int c0, int p0,
                          const half_t* __restrict__ s1, int c1c, int p1,
                          half_t* __restrict__ d, int ct) {
  long i = (long)blockIdx.x * blockDim.x + threadIdx.x;
  if (i >= (long)NP * ct) return;
  int c = (int)(i % ct);
  long np = i / ct;
  d[i] = (c < c0) ? s0[np * p0 + c] : s1[np * p1 + (c - c0)];
}

__global__ void k_rnet(const float* __restrict__ r, const float* __restrict__ net,
                       half_t* __restrict__ d) {
  long i = (long)blockIdx.x * blockDim.x + threadIdx.x;
  if (i < (long)NP * HD) d[i] = (half_t)(r[i] * net[i]);
}

__global__ void k_gru(const float* __restrict__ z, const float* __restrict__ q,
                      float* __restrict__ net_f, half_t* __restrict__ net128) {
  long i = (long)blockIdx.x * blockDim.x + threadIdx.x;
  if (i >= (long)NP * HD) return;
  float v = (1.f - z[i]) * net_f[i] + z[i] * q[i];
  net_f[i] = v;
  net128[i] = (half_t)v;
}

// c1 (channel-first) += dflow (channel-last, stride 16)
__global__ void k_addflow(const float* __restrict__ dflow, float* __restrict__ c1) {
  long i = (long)blockIdx.x * blockDim.x + threadIdx.x;
  if (i >= (long)NB * 2 * PP) return;
  int p = (int)(i % PP);
  long t = i / PP;
  int ch = (int)(t % 2);
  int n = (int)(t / 2);
  c1[i] += dflow[((long)n * PP + p) * 16 + ch];
}

// Convex 8x upsample of (c1 - coords0); mask channel-last [NP,576].
__global__ void k_upsample(const float* __restrict__ c1, const float* __restrict__ mask,
                           float* __restrict__ out) {
  const int OH = 8 * HH, OW = 8 * WW;
  long i = (long)blockIdx.x * blockDim.x + threadIdx.x;
  if (i >= (long)NB * 2 * OH * OW) return;
  int ox = (int)(i % OW);
  long t = i / OW;
  int oy = (int)(t % OH);
  t /= OH;
  int ch = (int)(t % 2);
  int n = (int)(t / 2);
  int iy = oy >> 3, ry = oy & 7, ix = ox >> 3, rx = ox & 7;
  const float* mp = mask + ((long)n * PP + iy * WW + ix) * 576 + (ry * 8 + rx);
  float lg[9], mx = -1e30f;
#pragma unroll
  for (int k = 0; k < 9; ++k) {
    lg[k] = mp[k * 64];
    mx = fmaxf(mx, lg[k]);
  }
  float se = 0.f;
#pragma unroll
  for (int k = 0; k < 9; ++k) { lg[k] = expf(lg[k] - mx); se += lg[k]; }
  float acc = 0.f;
#pragma unroll
  for (int k = 0; k < 9; ++k) {
    int yy = iy + k / 3 - 1, xx = ix + k % 3 - 1;
    float fv = 0.f;
    if (yy >= 0 && yy < HH && xx >= 0 && xx < WW) {
      float base = (ch == 0) ? (float)xx : (float)yy;
      fv = c1[((long)n * 2 + ch) * PP + yy * WW + xx] - base;
    }
    acc += lg[k] * 8.f * fv;
  }
  out[i] = acc / se;
}

// ------------------------------- host --------------------------------------

static inline int gridFor(long n, int b) { return (int)((n + b - 1) / b); }

extern "C" void kernel_launch(void* const* d_in, const int* in_sizes, int n_in,
                              void* d_out, int out_size, void* d_ws, size_t ws_size,
                              hipStream_t stream) {
  (void)in_sizes; (void)n_in; (void)out_size; (void)ws_size;

  const float* fmap1     = (const float*)d_in[0];
  const float* fmap2     = (const float*)d_in[1];
  const float* cnet      = (const float*)d_in[2];
  const float* flow_init = (const float*)d_in[3];

  // params in dict-insertion order: (W, b) pairs per spec.
  enum { Wc1, Wc2, Wf1, Wf2, Wme, Wz1, Wr1, Wq1, Wz2, Wr2, Wq2, Wfh1, Wfh2, Wm1, Wm2 };
  const int Wco[15]  = {256, 192, 128, 64, 126, 128, 128, 128, 128, 128, 128, 256, 2, 256, 576};
  const int Wci[15]  = {324, 256, 2, 128, 256, 384, 384, 384, 384, 384, 384, 128, 256, 128, 256};
  const int Wkh[15]  = {1, 3, 7, 3, 3, 1, 1, 1, 5, 5, 5, 3, 3, 3, 1};
  const int Wkw[15]  = {1, 3, 7, 3, 3, 5, 5, 5, 1, 1, 1, 3, 3, 3, 1};
  const int WpIn[15] = {352, 256, 32, 128, 256, 384, 384, 384, 384, 384, 384, 128, 256, 128, 256};
  const int WpOut[15]= {256, 192, 128, 64, 128, 128, 128, 128, 128, 128, 128, 256, 16, 256, 576};
  const float* wsrc[15];
  const float* bsrc[15];
  for (int t = 0; t < 15; ++t) {
    wsrc[t] = (const float*)d_in[4 + 2 * t];
    bsrc[t] = (const float*)d_in[5 + 2 * t];
  }

  // ---- workspace carve (recomputed identically each call) ----
  char* base = (char*)d_ws;
  size_t off = 0;
  auto alloc = [&](size_t bytes) -> void* {
    void* p = base + off;
    off = (off + bytes + 255) & ~(size_t)255;
    return p;
  };
  half_t* f1t  = (half_t*)alloc((size_t)NB * PP * CC * 2);
  half_t* f2t  = (half_t*)alloc((size_t)NB * PP * CC * 2);
  float* pyr0  = (float*)alloc((size_t)NP * PP * 4);
  float* pyr1  = (float*)alloc((size_t)NP * (PP / 4) * 4);
  float* pyr2  = (float*)alloc((size_t)NP * (PP / 16) * 4);
  float* pyr3  = (float*)alloc((size_t)NP * (PP / 64) * 4);
  half_t* wh[15];
  for (int t = 0; t < 15; ++t)
    wh[t] = (half_t*)alloc((size_t)WpOut[t] * Wkh[t] * Wkw[t] * WpIn[t] * 2);
  float*  net_f  = (float*)alloc((size_t)NP * HD * 4);
  half_t* net128 = (half_t*)alloc((size_t)NP * HD * 2);
  float*  c1     = (float*)alloc((size_t)NB * 2 * PP * 4);
  half_t* feat   = (half_t*)alloc((size_t)NP * FEATP * 2);
  half_t* flowh  = (half_t*)alloc((size_t)NP * 32 * 2);
  half_t* cor1   = (half_t*)alloc((size_t)NP * 256 * 2);
  half_t* cat_h  = (half_t*)alloc((size_t)NP * 256 * 2);  // [cor2(192) | flo2(64)]
  half_t* flo1   = (half_t*)alloc((size_t)NP * 128 * 2);
  half_t* xcat   = (half_t*)alloc((size_t)NP * 256 * 2);  // [inp(128)|mot(126)|flow(2)]
  half_t* hx_h   = (half_t*)alloc((size_t)NP * 384 * 2);
  half_t* rn_h   = (half_t*)alloc((size_t)NP * HD * 2);
  half_t* qin_h  = (half_t*)alloc((size_t)NP * 384 * 2);
  half_t* fh_h   = (half_t*)alloc((size_t)NP * 256 * 2);
  float* z_f     = (float*)alloc((size_t)NP * HD * 4);
  float* r_f     = (float*)alloc((size_t)NP * HD * 4);
  float* q_f     = (float*)alloc((size_t)NP * HD * 4);
  float* dflow   = (float*)alloc((size_t)NP * 16 * 4);
  float* mask_f  = (float*)alloc((size_t)NP * 576 * 4);

  const int TB = 256;

  // ---- one-time setup ----
  for (int t = 0; t < 15; ++t) {
    long n = (long)WpOut[t] * Wkh[t] * Wkw[t] * WpIn[t];
    k_wreorder<<<gridFor(n, TB), TB, 0, stream>>>(wsrc[t], wh[t], Wco[t], Wci[t],
                                                  Wkh[t], Wkw[t], WpIn[t], WpOut[t]);
  }
  {
    long n = (long)NB * CC * PP;
    k_transpose_pc<<<gridFor(n, TB), TB, 0, stream>>>(fmap1, f1t);
    k_transpose_pc<<<gridFor(n, TB), TB, 0, stream>>>(fmap2, f2t);
  }
  {
    int tiles = (PP / 16) * (PP / 16);  // 36864 tiles, 4 waves/block
    k_corr_wmma<<<dim3(tiles / 4, NB), 128, 0, stream>>>(f1t, f2t, pyr0, 1.0f / 16.0f);
  }
  k_avgpool2<<<gridFor((long)NP * (PP / 4), TB), TB, 0, stream>>>(pyr0, pyr1, NP, HH, WW);
  k_avgpool2<<<gridFor((long)NP * (PP / 16), TB), TB, 0, stream>>>(pyr1, pyr2, NP, HH / 2, WW / 2);
  k_avgpool2<<<gridFor((long)NP * (PP / 64), TB), TB, 0, stream>>>(pyr2, pyr3, NP, HH / 4, WW / 4);
  k_cnet_split<<<gridFor((long)NP * HD, TB), TB, 0, stream>>>(cnet, net_f, net128, xcat);
  k_init_c1<<<gridFor((long)NB * 2 * PP, TB), TB, 0, stream>>>(flow_init, c1);

  auto conv = [&](const half_t* in, int idx, float* of, half_t* oh,
                  int Cos, int coff, int act, float scale) {
    int blocks = ((WpOut[idx] / 16) * (NP / 16) + 3) / 4;
    k_conv_wmma<<<blocks, 128, 0, stream>>>(in, wh[idx], bsrc[idx], of, oh,
                                            WpIn[idx], Wco[idx], WpOut[idx],
                                            Wkh[idx], Wkw[idx],
                                            Wkh[idx] / 2, Wkw[idx] / 2,
                                            Cos, coff, act, scale);
  };

  float* outp = (float*)d_out;
  const long upsz = (long)NB * 2 * (8 * HH) * (8 * WW);
  const long e2 = (long)NB * 2 * PP;
  const long eh = (long)NP * HD;

  for (int it = 0; it < NITER; ++it) {
    // correlation lookup + flow (from start-of-iteration coords1)
    k_corr_lookup<<<gridFor((long)NP * FEATP, TB), TB, 0, stream>>>(pyr0, pyr1, pyr2, pyr3,
                                                                    c1, feat);
    k_flow_h<<<gridFor((long)NP * 32, TB), TB, 0, stream>>>(c1, flowh);
    // BasicMotionEncoder (cat and xcat assembled via epilogue channel offsets)
    conv(feat,  Wc1, nullptr, cor1,  256, 0,   1, 1.f);
    conv(cor1,  Wc2, nullptr, cat_h, 256, 0,   1, 1.f);   // cat[0..192)
    conv(flowh, Wf1, nullptr, flo1,  128, 0,   1, 1.f);
    conv(flo1,  Wf2, nullptr, cat_h, 256, 192, 1, 1.f);   // cat[192..256)
    conv(cat_h, Wme, nullptr, xcat,  256, 128, 1, 1.f);   // xcat[128..256) (126+2 pad)
    k_flowpair<<<gridFor((long)NP * 2, TB), TB, 0, stream>>>(c1, xcat);  // xcat[254..256)
    // SepConvGRU horizontal (1x5)
    k_concat2<<<gridFor((long)NP * 384, TB), TB, 0, stream>>>(net128, 128, 128,
                                                              xcat, 256, 256, hx_h, 384);
    conv(hx_h, Wz1, z_f, nullptr, 128, 0, 2, 1.f);
    conv(hx_h, Wr1, r_f, nullptr, 128, 0, 2, 1.f);
    k_rnet<<<gridFor(eh, TB), TB, 0, stream>>>(r_f, net_f, rn_h);
    k_concat2<<<gridFor((long)NP * 384, TB), TB, 0, stream>>>(rn_h, 128, 128,
                                                              xcat, 256, 256, qin_h, 384);
    conv(qin_h, Wq1, q_f, nullptr, 128, 0, 3, 1.f);
    k_gru<<<gridFor(eh, TB), TB, 0, stream>>>(z_f, q_f, net_f, net128);
    // SepConvGRU vertical (5x1)
    k_concat2<<<gridFor((long)NP * 384, TB), TB, 0, stream>>>(net128, 128, 128,
                                                              xcat, 256, 256, hx_h, 384);
    conv(hx_h, Wz2, z_f, nullptr, 128, 0, 2, 1.f);
    conv(hx_h, Wr2, r_f, nullptr, 128, 0, 2, 1.f);
    k_rnet<<<gridFor(eh, TB), TB, 0, stream>>>(r_f, net_f, rn_h);
    k_concat2<<<gridFor((long)NP * 384, TB), TB, 0, stream>>>(rn_h, 128, 128,
                                                              xcat, 256, 256, qin_h, 384);
    conv(qin_h, Wq2, q_f, nullptr, 128, 0, 3, 1.f);
    k_gru<<<gridFor(eh, TB), TB, 0, stream>>>(z_f, q_f, net_f, net128);
    // flow head + mask head
    conv(net128, Wfh1, nullptr, fh_h, 256, 0, 1, 1.f);
    conv(fh_h,   Wfh2, dflow, nullptr, 16, 0, 0, 1.f);
    conv(net128, Wm1, nullptr, fh_h, 256, 0, 1, 1.f);
    conv(fh_h,   Wm2, mask_f, nullptr, 576, 0, 0, 0.25f);
    // coords update + convex upsample of (c1_new - coords0)
    k_addflow<<<gridFor(e2, TB), TB, 0, stream>>>(dflow, c1);
    k_upsample<<<gridFor(upsz, TB), TB, 0, stream>>>(c1, mask_f, outp + (long)it * upsz);
  }
}
